// Attention_11347303596728
// MI455X (gfx1250) — matrix operand order repaired and checked by reading, not compiled, not run on hardware
//
#include <hip/hip_runtime.h>
#include <hip/hip_bf16.h>

// ---------------------------------------------------------------------------
// CDNA5 / gfx1250 attention prefill, bf16-WMMA everywhere.
//   out = softmax(causal + (rms(rope(xWq)) . rms(rope(xWk))^T)/sqrt(hd)) @ V @ Wo
// Activations (Q, K^T, V, ctx) stored bf16; GEMMs use v_wmma_f32_16x16x32_bf16
// with f32 accumulation, double-buffered LDS, register prefetch of the next
// K-slice (one barrier per slice), packed-dword LDS staging.
// ---------------------------------------------------------------------------

typedef __bf16        v16bf  __attribute__((ext_vector_type(16)));
typedef __bf16        bf16x2 __attribute__((ext_vector_type(2)));
typedef float         v8f    __attribute__((ext_vector_type(8)));
typedef unsigned int  v8u    __attribute__((ext_vector_type(8)));

namespace {
constexpr int B_   = 2;
constexpr int S_   = 1024;
constexpr int D_   = 4096;
constexpr int H_   = 32;
constexpr int KVH_ = 8;
constexpr int HD_  = 128;
constexpr int NREP_ = H_ / KVH_;          // 4

constexpr int BM = 128, BN = 128, BK = 32;
constexpr int BKW = BK / 2 + 1;           // dwords per LDS row (stride 17 -> conflict free)
constexpr float QK_SCALE = 0.08838834764831845f;   // 1/sqrt(128)
}

// Packed bf16-pair builders: float path should lower to v_cvt_pk_bf16_f32.
__device__ __forceinline__ unsigned pack2(float a, float b) {
  bf16x2 p; p[0] = (__bf16)a; p[1] = (__bf16)b;
  return __builtin_bit_cast(unsigned, p);
}
__device__ __forceinline__ unsigned pack2(__bf16 a, __bf16 b) {
  bf16x2 p; p[0] = a; p[1] = b;
  return __builtin_bit_cast(unsigned, p);
}

__device__ __forceinline__ void store_elt(float*  p, float v) { *p = v; }
__device__ __forceinline__ void store_elt(__bf16* p, float v) { *p = (__bf16)v; }

// ---------------------------------------------------------------------------
// 128x128-tile GEMM: C = alpha * A(MxK, lda) * B(KxN, ldb, row-major).
// 256 threads = 8 waves (4 along M x 2 along N); each wave owns a 32x64 patch
// = 2x4 grid of 16x16x32 WMMA tiles.
// ---------------------------------------------------------------------------
template <typename TA, typename TB, typename TC>
__device__ __forceinline__ void gemm_wmma_128x128(
    const TA* __restrict__ A, long lda,
    const TB* __restrict__ Bp, long ldb,
    TC* __restrict__ C, long ldc,
    int mBase, int nBase, int kEnd, float alpha)
{
  __shared__ unsigned As32[2][BM][BKW];   // As32[buf][m][w] : K = 2w, 2w+1
  __shared__ unsigned Bs32[2][BN][BKW];   // Bs32[buf][n][w]

  const int tid  = threadIdx.x;
  const int lane = tid & 31;
  const int wave = tid >> 5;
  const int wm   = wave & 3;      // 4 waves along M
  const int wn   = wave >> 2;     // 2 waves along N
  const int half = lane >> 4;
  const int l16  = lane & 15;

  const v8f vzero = {0.f, 0.f, 0.f, 0.f, 0.f, 0.f, 0.f, 0.f};
  v8f acc[2][4];
#pragma unroll
  for (int mt = 0; mt < 2; ++mt)
#pragma unroll
    for (int nt = 0; nt < 4; ++nt)
      acc[mt][nt] = vzero;

  // --- staging assignments (256 threads) ---
  // A: each thread covers 16 consecutive k of one row (bf16 pairs are
  //    memory-adjacent -> direct dword loads when TA is bf16).
  const int am = tid >> 1;              // row 0..127
  const int ak = (tid & 1) << 4;        // k base 0 or 16
  // B: each thread loads 8 consecutive n of two adjacent k rows, then packs
  //    (k, k+1) pairs per n into dwords of the transposed LDS tile.
  const int bkp = (tid >> 4) << 1;      // k-pair base: 0,2,...,30
  const int bng = (tid & 15) << 3;      // n base: 0,8,...,120

  float    aregf[16];                   // TA == float path
  unsigned awords[8];                   // TA == bf16 path (raw dwords)
  TB breg0[8], breg1[8];

  auto load_slice = [&](int kb) {
    if constexpr (__is_same(TA, __bf16)) {
      const unsigned* ap =
          (const unsigned*)(A + (long)(mBase + am) * lda + (kb + ak));
#pragma unroll
      for (int i = 0; i < 8; ++i) awords[i] = ap[i];
    } else {
      const TA* ap = A + (long)(mBase + am) * lda + (kb + ak);
#pragma unroll
      for (int i = 0; i < 16; ++i) aregf[i] = ap[i];
    }
    const TB* bp0 = Bp + (long)(kb + bkp) * ldb + (nBase + bng);
    const TB* bp1 = bp0 + ldb;
#pragma unroll
    for (int i = 0; i < 8; ++i) { breg0[i] = bp0[i]; breg1[i] = bp1[i]; }
  };

  auto store_slice = [&](int bufi) {
    if constexpr (__is_same(TA, __bf16)) {
#pragma unroll
      for (int i = 0; i < 8; ++i)
        As32[bufi][am][(ak >> 1) + i] = awords[i];
    } else {
#pragma unroll
      for (int i = 0; i < 8; ++i)
        As32[bufi][am][(ak >> 1) + i] = pack2(aregf[2 * i], aregf[2 * i + 1]);
    }
#pragma unroll
    for (int i = 0; i < 8; ++i)
      Bs32[bufi][bng + i][bkp >> 1] = pack2(breg0[i], breg1[i]);
  };

  load_slice(0);
  store_slice(0);
  __syncthreads();

  int buf = 0;
  for (int kb = 0; kb < kEnd; kb += BK) {
    const bool has_next = (kb + BK) < kEnd;
    if (has_next) load_slice(kb + BK);   // prefetch next slice into registers

    // --- fragments from LDS[buf] ---
    // A (16-bit 16x32, ISA 7.12.2): lane m=l16; vgpr v holds
    //   k = (v/4)*16 + half*8 + (v%4)*2  ->  word w = (v/4)*8 + half*4 + (v%4)
    v8u araw[2];
#pragma unroll
    for (int mt = 0; mt < 2; ++mt) {
      const unsigned* arow = &As32[buf][wm * 32 + mt * 16 + l16][0];
#pragma unroll
      for (int v = 0; v < 8; ++v)
        araw[mt][v] = arow[((v >> 2) << 3) + (half << 2) + (v & 3)];
    }
    // B (16-bit KxN): lane n=l16; same k order as A:
    //   vgpr v holds k = (v/4)*16 + half*8 + (v%4)*2  ->  word w = (v/4)*8 + half*4 + (v%4)
    v8u braw[4];
#pragma unroll
    for (int nt = 0; nt < 4; ++nt) {
      const unsigned* brow = &Bs32[buf][wn * 64 + nt * 16 + l16][0];
#pragma unroll
      for (int v = 0; v < 8; ++v)
        braw[nt][v] = brow[((v >> 2) << 3) + (half << 2) + (v & 3)];
    }

#pragma unroll
    for (int mt = 0; mt < 2; ++mt)
#pragma unroll
      for (int nt = 0; nt < 4; ++nt)
        acc[mt][nt] = __builtin_amdgcn_wmma_f32_16x16x32_bf16(
            false, __builtin_bit_cast(v16bf, araw[mt]),
            false, __builtin_bit_cast(v16bf, braw[nt]),
            (short)0, acc[mt][nt], false, false);

    if (has_next) {
      store_slice(buf ^ 1);   // other buffer: no race with current readers
      __syncthreads();        // one barrier per K-slice
    }
    buf ^= 1;
  }

  // C/D layout: VGPR r -> M = r + 8*half, N = l16
#pragma unroll
  for (int mt = 0; mt < 2; ++mt)
#pragma unroll
    for (int nt = 0; nt < 4; ++nt)
#pragma unroll
      for (int r = 0; r < 8; ++r) {
        const int row = mBase + wm * 32 + mt * 16 + r + (half << 3);
        const int col = nBase + wn * 64 + nt * 16 + l16;
        store_elt(C + (long)row * ldc + col, acc[mt][nt][r] * alpha);
      }
}

// ---------------------------------------------------------------------------
// GEMM wrapper kernels
// ---------------------------------------------------------------------------

// QKV projections: f32 x @ f32 w -> bf16 activations
__global__ __launch_bounds__(256)
void k_gemm_qkv(const float* __restrict__ A, const float* __restrict__ Bw,
                __bf16* __restrict__ C, long lda, long ldb, long ldc, int K)
{
  gemm_wmma_128x128<float, float, __bf16>(A, lda, Bw, ldb, C, ldc,
                                          blockIdx.y * BM, blockIdx.x * BN, K, 1.0f);
}

// scores: bf16 Q (B,S,H*HD) x bf16 K^T (B,KVH,HD,S) -> f32 raw scores
__global__ __launch_bounds__(256)
void k_scores(const __bf16* __restrict__ Qb, const __bf16* __restrict__ Kt,
              float* __restrict__ attn)
{
  const int bh = blockIdx.z;
  const int b = bh / H_, h = bh % H_, kvh = h / NREP_;
  const int nBase = blockIdx.x * BN;
  const int mBase = blockIdx.y * BM;
  if (nBase > mBase + (BM - 1)) return;   // fully-masked tile: skip

  const __bf16* Ab = Qb + (long)b * S_ * (H_ * HD_) + (long)h * HD_;
  const __bf16* Bb = Kt + ((long)(b * KVH_ + kvh) * HD_) * S_;   // [d][s], ldb = S
  float* Cb = attn + (long)bh * S_ * S_;
  gemm_wmma_128x128<__bf16, __bf16, float>(Ab, H_ * HD_, Bb, S_, Cb, S_,
                                           mBase, nBase, HD_, 1.0f);
}

// ctx: f32 attn (S x S) x bf16 V -> bf16 ctx (B,S,H*HD); causal K-limit
__global__ __launch_bounds__(256)
void k_ctx(const float* __restrict__ attn, const __bf16* __restrict__ Vb,
           __bf16* __restrict__ Cx)
{
  const int bh = blockIdx.z;
  const int b = bh / H_, h = bh % H_, kvh = h / NREP_;
  const int mBase = blockIdx.y * BM;        // N = HD = 128 = one block column
  const float* Ab = attn + (long)bh * S_ * S_;
  const __bf16* Bb = Vb + (long)b * S_ * (KVH_ * HD_) + (long)kvh * HD_;
  __bf16* Cb = Cx + (long)b * S_ * (H_ * HD_) + (long)h * HD_;
  gemm_wmma_128x128<float, __bf16, __bf16>(Ab, S_, Bb, KVH_ * HD_, Cb, H_ * HD_,
                                           mBase, 0, mBase + BM, 1.0f);
}

// out: bf16 ctx x f32 wo -> f32 out
__global__ __launch_bounds__(256)
void k_gemm_out(const __bf16* __restrict__ A, const float* __restrict__ Bw,
                float* __restrict__ C, long lda, long ldb, long ldc, int K)
{
  gemm_wmma_128x128<__bf16, float, float>(A, lda, Bw, ldb, C, ldc,
                                          blockIdx.y * BM, blockIdx.x * BN, K, 1.0f);
}

// ---------------------------------------------------------------------------
// RoPE (interleaved pairs) + RMS-norm over HD=128.  Q updated in place (bf16);
// K written TRANSPOSED into Kt[b][kvh][d][s] so the scores GEMM reads its
// B operand contiguously in N.  One wave per (b,s,head) row; lane j handles
// pairs j and j+32; wave-shuffle reduction (wave32).
// ---------------------------------------------------------------------------
__global__ __launch_bounds__(128)
void k_rope_qknorm(__bf16* __restrict__ Qb, const __bf16* __restrict__ Kb,
                   __bf16* __restrict__ Kt,
                   const float* __restrict__ fcos, const float* __restrict__ fsin)
{
  const int lane = threadIdx.x & 31;
  const int wave = threadIdx.x >> 5;
  const long row = (long)blockIdx.x * 4 + wave;   // [0, B*S*(H+KVH))
  const int HT = H_ + KVH_;
  const int hh = (int)(row % HT);
  const long bs = row / HT;                       // b*S + s
  const int s = (int)(bs % S_);
  const int b = (int)(bs / S_);

  const bool isQ = hh < H_;
  const int kv = hh - H_;
  const __bf16* src = isQ ? (Qb + (bs * H_ + hh) * HD_)
                          : (Kb + (bs * KVH_ + kv) * HD_);

  const int j0 = lane, j1 = lane + 32;
  const float c0 = fcos[s * 64 + j0], sn0 = fsin[s * 64 + j0];
  const float c1 = fcos[s * 64 + j1], sn1 = fsin[s * 64 + j1];
  const float re0 = (float)src[2 * j0], im0 = (float)src[2 * j0 + 1];
  const float re1 = (float)src[2 * j1], im1 = (float)src[2 * j1 + 1];
  const float oRe0 = re0 * c0 - im0 * sn0, oIm0 = re0 * sn0 + im0 * c0;
  const float oRe1 = re1 * c1 - im1 * sn1, oIm1 = re1 * sn1 + im1 * c1;

  float ss = oRe0 * oRe0 + oIm0 * oIm0 + oRe1 * oRe1 + oIm1 * oIm1;
#pragma unroll
  for (int off = 16; off; off >>= 1) ss += __shfl_xor(ss, off, 32);
  const float r = rsqrtf(ss * (1.0f / 128.0f) + 1e-5f);

  if (isQ) {
    __bf16* q = Qb + (bs * H_ + hh) * HD_;
    q[2 * j0]     = (__bf16)(oRe0 * r);
    q[2 * j0 + 1] = (__bf16)(oIm0 * r);
    q[2 * j1]     = (__bf16)(oRe1 * r);
    q[2 * j1 + 1] = (__bf16)(oIm1 * r);
  } else {
    __bf16* kt = Kt + ((long)(b * KVH_ + kv) * HD_) * S_ + s;   // [d][s]
    kt[(long)(2 * j0) * S_]     = (__bf16)(oRe0 * r);
    kt[(long)(2 * j0 + 1) * S_] = (__bf16)(oIm0 * r);
    kt[(long)(2 * j1) * S_]     = (__bf16)(oRe1 * r);
    kt[(long)(2 * j1 + 1) * S_] = (__bf16)(oIm1 * r);
  }
}

// ---------------------------------------------------------------------------
// Causal softmax, in place over attn rows (length S=1024).  One wave per row;
// the row lives in 32 registers per lane.  Masked entries written as exact 0
// (reference: exp(-1e9 - max) underflows to 0 in f32).  Scale applied here.
// ---------------------------------------------------------------------------
__global__ __launch_bounds__(256)
void k_softmax(float* __restrict__ attn)
{
  const int lane = threadIdx.x & 31;
  const int wave = threadIdx.x >> 5;
  const long row = (long)blockIdx.x * 8 + wave;     // [0, B*H*S)
  const int i = (int)(row % S_);                    // query position
  float* p = attn + row * (long)S_;

  float vals[32];
  float m = -3.0e38f;
#pragma unroll
  for (int it = 0; it < 32; ++it) {
    const int j = it * 32 + lane;
    const float v = (j <= i) ? p[j] * QK_SCALE : -3.0e38f;
    vals[it] = v;
    m = fmaxf(m, v);
  }
#pragma unroll
  for (int off = 16; off; off >>= 1) m = fmaxf(m, __shfl_xor(m, off, 32));

  float sum = 0.f;
#pragma unroll
  for (int it = 0; it < 32; ++it) {
    const float e = (vals[it] > -1.0e38f) ? __expf(vals[it] - m) : 0.f;
    vals[it] = e;
    sum += e;
  }
#pragma unroll
  for (int off = 16; off; off >>= 1) sum += __shfl_xor(sum, off, 32);
  const float inv = 1.0f / sum;

#pragma unroll
  for (int it = 0; it < 32; ++it)
    p[it * 32 + lane] = vals[it] * inv;
}

// ---------------------------------------------------------------------------
extern "C" void kernel_launch(void* const* d_in, const int* in_sizes, int n_in,
                              void* d_out, int out_size, void* d_ws, size_t ws_size,
                              hipStream_t stream)
{
  (void)in_sizes; (void)n_in; (void)out_size; (void)ws_size;

  const float* x    = (const float*)d_in[0];
  const float* wq   = (const float*)d_in[1];
  const float* wk   = (const float*)d_in[2];
  const float* wv   = (const float*)d_in[3];
  const float* wo   = (const float*)d_in[4];
  const float* fcos = (const float*)d_in[5];
  const float* fsin = (const float*)d_in[6];
  // d_in[7] (mask), d_in[8] (start_pos = 0): causal prefill handled analytically.

  float* out  = (float*)d_out;                          // (B,S,D) f32
  float* attn = out + (size_t)B_ * S_ * D_;             // (B,H,S,S) f32

  // workspace carve-up (bf16): Q 16MB | K 4MB | Kt 4MB | V 4MB | ctx 16MB
  __bf16* Qb = (__bf16*)d_ws;
  __bf16* Kb = Qb + (size_t)(B_ * S_) * (H_ * HD_);
  __bf16* Kt = Kb + (size_t)(B_ * S_) * (KVH_ * HD_);
  __bf16* Vb = Kt + (size_t)(B_ * KVH_) * HD_ * S_;
  __bf16* Cx = Vb + (size_t)(B_ * S_) * (KVH_ * HD_);

  const int M = B_ * S_;            // 2048
  dim3 blk(256);

  // 1) QKV projections (f32 in, bf16 out)
  k_gemm_qkv<<<dim3((H_ * HD_) / BN, M / BM), blk, 0, stream>>>(
      x, wq, Qb, D_, (long)(H_ * HD_), (long)(H_ * HD_), D_);
  k_gemm_qkv<<<dim3((KVH_ * HD_) / BN, M / BM), blk, 0, stream>>>(
      x, wk, Kb, D_, (long)(KVH_ * HD_), (long)(KVH_ * HD_), D_);
  k_gemm_qkv<<<dim3((KVH_ * HD_) / BN, M / BM), blk, 0, stream>>>(
      x, wv, Vb, D_, (long)(KVH_ * HD_), (long)(KVH_ * HD_), D_);

  // 2) RoPE + QK RMS-norm; Q in place, K -> transposed Kt
  k_rope_qknorm<<<(B_ * S_ * (H_ + KVH_)) / 4, 128, 0, stream>>>(
      Qb, Kb, Kt, fcos, fsin);

  // 3) raw scores into the attn output region (softmax overwrites in place)
  k_scores<<<dim3(S_ / BN, S_ / BM, B_ * H_), blk, 0, stream>>>(Qb, Kt, attn);

  // 4) causal softmax (scale folded in), in place
  k_softmax<<<(B_ * H_ * S_) / 8, 256, 0, stream>>>(attn);

  // 5) ctx = attn @ V  (bf16, stored (B,S,H*HD))
  k_ctx<<<dim3(1, S_ / BM, B_ * H_), blk, 0, stream>>>(attn, Vb, Cx);

  // 6) out = ctx @ wo  (f32 result)
  k_gemm_out<<<dim3(D_ / BN, M / BM), blk, 0, stream>>>(
      Cx, wo, out, (long)(H_ * HD_), D_, D_, H_ * HD_);
}
